// Lfm2MoeAttentionDecoderLayer_2113123909697
// MI455X (gfx1250) — compile-verified, matrix operand order checked by reading, not run on hardware
//
#include <hip/hip_runtime.h>
#include <cstdint>
#include <cstddef>

#define T_TOK 1024
#define HID 2048
#define N_HEADS 32
#define N_KV 8
#define HEAD_DIM 64
#define QKV_N 3072
#define N_EXP 32
#define TOP_K 4
#define FF 1792
#define CAP 256
#define EPSF 1e-5f
#define ATT_SCALE 0.125f

typedef __attribute__((ext_vector_type(16))) __bf16 bf16x16;
typedef __attribute__((ext_vector_type(8)))  __bf16 bf16x8;
typedef __attribute__((ext_vector_type(2)))  __bf16 bf16x2;
typedef __attribute__((ext_vector_type(8)))  float  f32x8;

#define WMMA_BF16(a, b, c) \
  __builtin_amdgcn_wmma_f32_16x16x32_bf16(false, (a), false, (b), (short)0, (c), false, false)

// ---------------------------------------------------------------------------
// Fragment helpers.
// A-fragment (16x32 bf16): lane (m=lane&15, h=lane>>4), element e maps to
// k = k0 + 16*(e/8) + 8*h + (e%8)  -> two contiguous 16-byte chunks of a row.
// B-fragment (32x16 bf16) has the identical per-lane k pattern with lane&15 = n.
// ---------------------------------------------------------------------------
__device__ __forceinline__ bf16x16 load_frag(const __bf16* rowp, int k0, int h) {
  const __bf16* p = rowp + k0 + h * 8;
  bf16x8 lo = *(const bf16x8*)(p);
  bf16x8 hi = *(const bf16x8*)(p + 16);
  bf16x16 r;
#pragma unroll
  for (int i = 0; i < 8; ++i) { r[i] = lo[i]; r[i + 8] = hi[i]; }
  return r;
}

// LDS B tile stored transposed: sB[col][k], col stride 48 bf16 (96B, 16B aligned)
__device__ __forceinline__ bf16x16 read_b_lds(const __bf16* sB, int col, int h) {
  return load_frag(sB + col * 48, 0, h);
}

// ---------------------------------------------------------------------------
// Register-pipelined B staging.  Thread cell mapping is affine & uniform:
//   col = tid & 127 (constant per thread),  k rows = (tid>>7)*2 + 4i and +1.
// With a compile-time LDB the loads become saddr-base + one 32-bit voffset
// + immediate ioffsets (4*LDB*i fits the signed 24-bit field), eliminating
// per-element 64-bit address math.  stage_store packs pairs with
// v_cvt_pk_bf16_f32 into one aligned ds_store_b32 per cell.
// ---------------------------------------------------------------------------
struct StageRegs { float lo[8]; float hi[8]; };

// Bk = B + (size_t)k0*LDB + n0 (uniform); off0 = (tid>>7)*2*LDB + (tid&127)
template <int LDB>
__device__ __forceinline__ void stage_load(StageRegs& r, const float* __restrict__ Bk,
                                           int off0) {
#pragma unroll
  for (int i = 0; i < 8; ++i) {
    r.lo[i] = Bk[off0 + 4 * LDB * i];
    r.hi[i] = Bk[off0 + 4 * LDB * i + LDB];
  }
}

__device__ __forceinline__ void stage_store(const StageRegs& r, __bf16* sB, int tid) {
  __bf16* p = sB + (tid & 127) * 48 + (tid >> 7) * 2;
#pragma unroll
  for (int i = 0; i < 8; ++i) {
    bf16x2 pk;
    pk[0] = (__bf16)r.lo[i];
    pk[1] = (__bf16)r.hi[i];
    *(bf16x2*)(p + 4 * i) = pk;   // 8-byte immediate offsets, 4B aligned
  }
}

__device__ __forceinline__ float wave_sum(float v) {
#pragma unroll
  for (int o = 16; o > 0; o >>= 1) v += __shfl_xor(v, o, 32);
  return v;
}

// ---------------------------------------------------------------------------
// x = a + b ; res_out = x ; h = rmsnorm(x)*w  (bf16 and optional fp32 copies)
// ---------------------------------------------------------------------------
__global__ __launch_bounds__(256) void k_addnorm(const float* __restrict__ a,
                                                 const float* __restrict__ b,
                                                 const float* __restrict__ w,
                                                 float* __restrict__ res_out,
                                                 __bf16* __restrict__ h_bf,
                                                 float* __restrict__ h_f32) {
  const int t = blockIdx.x;
  const int tid = threadIdx.x;
  const int wave = tid >> 5, lane = tid & 31;
  const float* ra = a + (size_t)t * HID;
  const float* rb = b + (size_t)t * HID;
  __shared__ float red[8];
  float v[8];
  float ss = 0.f;
#pragma unroll
  for (int i = 0; i < 8; ++i) {
    int c = tid + 256 * i;
    float x = ra[c] + rb[c];
    v[i] = x;
    ss += x * x;
    res_out[(size_t)t * HID + c] = x;
  }
  ss = wave_sum(ss);
  if (lane == 0) red[wave] = ss;
  __syncthreads();
  float tot = 0.f;
#pragma unroll
  for (int i = 0; i < 8; ++i) tot += red[i];
  float r = rsqrtf(tot / (float)HID + EPSF);
#pragma unroll
  for (int i = 0; i < 8; ++i) {
    int c = tid + 256 * i;
    float hn = v[i] * r * w[c];
    h_bf[(size_t)t * HID + c] = (__bf16)hn;
    if (h_f32) h_f32[(size_t)t * HID + c] = hn;
  }
}

// ---------------------------------------------------------------------------
// Generic GEMM: C[M,N] f32 = A[M,K] bf16 (row major) x B[K,N] f32 (row major)
// Block tile 64(M) x 128(N), K-step 32, 8 waves (4 M-tiles x 2 N-halves),
// each wave computes 16x64 via four 16x16x32 WMMA accumulators.
// Staging is register-pipelined: tile k+1's global loads overlap tile k's WMMAs.
// ---------------------------------------------------------------------------
template <int N, int K>
__global__ __launch_bounds__(256) void k_gemm(const __bf16* __restrict__ A,
                                              const float* __restrict__ B,
                                              float* __restrict__ C) {
  __shared__ alignas(16) __bf16 sB[128 * 48];
  const int tid = threadIdx.x;
  const int wave = tid >> 5, lane = tid & 31;
  const int h = lane >> 4, ln = lane & 15;
  const int m0 = blockIdx.y * 64;
  const int n0 = blockIdx.x * 128;
  const int wm = (wave & 3) * 16;
  const int wn = (wave >> 2) * 64;
  const __bf16* Ap = A + (size_t)(m0 + wm + ln) * K;
  const float* Bk = B + n0;
  const int off0 = (tid >> 7) * 2 * N + (tid & 127);
  StageRegs sr;
  stage_load<N>(sr, Bk, off0);
  f32x8 acc[4] = {};
  for (int k0 = 0; k0 < K; k0 += 32) {
    __syncthreads();
    stage_store(sr, sB, tid);
    __syncthreads();
    Bk += 32 * N;
    if (k0 + 32 < K) stage_load<N>(sr, Bk, off0);  // in flight over WMMAs
    bf16x16 af = load_frag(Ap, 0, h);
    Ap += 32;
#pragma unroll
    for (int ns = 0; ns < 4; ++ns) {
      bf16x16 bf = read_b_lds(sB, wn + ns * 16 + ln, h);
      acc[ns] = WMMA_BF16(af, bf, acc[ns]);
    }
  }
#pragma unroll
  for (int ns = 0; ns < 4; ++ns) {
    int col = n0 + wn + ns * 16 + ln;
#pragma unroll
    for (int r = 0; r < 8; ++r) {
      int row = m0 + wm + r + 8 * h;
      C[(size_t)row * N + col] = acc[ns][r];
    }
  }
}

// ---------------------------------------------------------------------------
// Per-head QK RMSNorm + RoPE; V transposed to Vt[kvh][d][t] (bf16).
// One wave handles one head (lane l owns dims l and l+32 — the RoPE pair).
// ---------------------------------------------------------------------------
__global__ __launch_bounds__(256) void k_rope(const float* __restrict__ qkv,
                                              const float* __restrict__ qw,
                                              const float* __restrict__ kw,
                                              const int* __restrict__ positions,
                                              __bf16* __restrict__ Qo,
                                              __bf16* __restrict__ Ko,
                                              __bf16* __restrict__ Vto) {
  const int t = blockIdx.x;
  const int wave = threadIdx.x >> 5, lane = threadIdx.x & 31;
  const float pos = (float)positions[t];
  // inv_freq = THETA^(-lane/32), THETA = 1e6 -> ln = 13.815510558
  const float inv_freq = __expf(-(float)lane * (13.815510558f / 32.f));
  const float ang = pos * inv_freq;
  const float c = __cosf(ang), s = __sinf(ang);
  const float* base = qkv + (size_t)t * QKV_N;

  for (int hh = wave; hh < N_HEADS; hh += 8) {     // Q heads
    const float* x = base + hh * HEAD_DIM;
    float x1 = x[lane], x2 = x[lane + 32];
    float ss = wave_sum(x1 * x1 + x2 * x2);
    float r = rsqrtf(ss / (float)HEAD_DIM + EPSF);
    float n1 = x1 * r * qw[lane], n2 = x2 * r * qw[lane + 32];
    __bf16* row = Qo + ((size_t)hh * T_TOK + t) * HEAD_DIM;
    row[lane]      = (__bf16)(n1 * c - n2 * s);
    row[lane + 32] = (__bf16)(n2 * c + n1 * s);
  }
  if (wave < N_KV) {                                // K head + V transpose
    const float* x = base + (N_HEADS * HEAD_DIM) + wave * HEAD_DIM;
    float x1 = x[lane], x2 = x[lane + 32];
    float ss = wave_sum(x1 * x1 + x2 * x2);
    float r = rsqrtf(ss / (float)HEAD_DIM + EPSF);
    float n1 = x1 * r * kw[lane], n2 = x2 * r * kw[lane + 32];
    __bf16* row = Ko + ((size_t)wave * T_TOK + t) * HEAD_DIM;
    row[lane]      = (__bf16)(n1 * c - n2 * s);
    row[lane + 32] = (__bf16)(n2 * c + n1 * s);

    const float* xv = base + (N_HEADS + N_KV) * HEAD_DIM + wave * HEAD_DIM;
    Vto[((size_t)wave * HEAD_DIM + lane) * T_TOK + t]      = (__bf16)xv[lane];
    Vto[((size_t)wave * HEAD_DIM + lane + 32) * T_TOK + t] = (__bf16)xv[lane + 32];
  }
}

// ---------------------------------------------------------------------------
// Causal flash attention, one wave per (head, 16-query tile).
// Computes S^T = K*Q^T so that the WMMA C-layout of P^T is directly the
// B-operand layout for O^T += V^T * P^T (no cross-lane shuffles needed).
// ---------------------------------------------------------------------------
__global__ __launch_bounds__(256) void k_attn(const __bf16* __restrict__ Q,
                                              const __bf16* __restrict__ Kc,
                                              const __bf16* __restrict__ Vt,
                                              __bf16* __restrict__ Aout) {
  const int head = blockIdx.y;
  const int wave = threadIdx.x >> 5, lane = threadIdx.x & 31;
  const int h = lane >> 4, ln = lane & 15;
  const int q0 = (blockIdx.x * 8 + wave) * 16;
  const int kvh = head >> 2;  // rep = 4
  const __bf16* Qh = Q + ((size_t)head * T_TOK + q0) * HEAD_DIM;
  const __bf16* Kh = Kc + (size_t)kvh * T_TOK * HEAD_DIM;
  const __bf16* Vh = Vt + (size_t)kvh * HEAD_DIM * T_TOK;

  // Q^T as B fragments (lane&15 = query column), dk = 32-wide d chunks
  bf16x16 qb[2];
#pragma unroll
  for (int dk = 0; dk < 2; ++dk) qb[dk] = load_frag(Qh + (size_t)ln * HEAD_DIM, dk * 32, h);

  float mrow = -1e30f, lrow = 0.f;
  f32x8 ot[4] = {};  // O^T tiles: d = dm*16 + r + 8h, col q = ln
  const int q = q0 + ln;
  const int kend = q0 + 16;

  for (int kb = 0; kb < kend; kb += 32) {
    f32x8 st[2] = {};
#pragma unroll
    for (int ns = 0; ns < 2; ++ns)
#pragma unroll
      for (int dk = 0; dk < 2; ++dk) {
        bf16x16 ka = load_frag(Kh + (size_t)(kb + ns * 16 + ln) * HEAD_DIM, dk * 32, h);
        st[ns] = WMMA_BF16(ka, qb[dk], st[ns]);
      }
    // scale + causal mask + local max
    float cmax = -1e30f;
#pragma unroll
    for (int ns = 0; ns < 2; ++ns)
#pragma unroll
      for (int r = 0; r < 8; ++r) {
        int key = kb + ns * 16 + r + 8 * h;
        float v = st[ns][r] * ATT_SCALE;
        v = (key <= q) ? v : -1e30f;
        st[ns][r] = v;
        cmax = fmaxf(cmax, v);
      }
    cmax = fmaxf(cmax, __shfl_xor(cmax, 16, 32));   // combine both key halves
    float newm = fmaxf(mrow, cmax);
    float alpha = __expf(mrow - newm);
    float psum = 0.f;
    bf16x16 pb;   // P^T as B fragment: element e = ns*8 + r  ->  k = ns*16 + 8h + r
#pragma unroll
    for (int ns = 0; ns < 2; ++ns)
#pragma unroll
      for (int r = 0; r < 8; ++r) {
        float p = __expf(st[ns][r] - newm);
        psum += p;
        pb[ns * 8 + r] = (__bf16)p;
      }
    psum += __shfl_xor(psum, 16, 32);
    lrow = lrow * alpha + psum;
    mrow = newm;
#pragma unroll
    for (int dm = 0; dm < 4; ++dm) {
#pragma unroll
      for (int r = 0; r < 8; ++r) ot[dm][r] *= alpha;
      bf16x16 va = load_frag(Vh + (size_t)(dm * 16 + ln) * T_TOK, kb, h);
      ot[dm] = WMMA_BF16(va, pb, ot[dm]);
    }
  }
  const float inv_l = 1.0f / lrow;
#pragma unroll
  for (int dm = 0; dm < 4; ++dm)
#pragma unroll
    for (int r = 0; r < 8; ++r) {
      int d = dm * 16 + r + 8 * h;
      Aout[(size_t)(q0 + ln) * HID + head * HEAD_DIM + d] = (__bf16)(ot[dm][r] * inv_l);
    }
}

// ---------------------------------------------------------------------------
// MoE gate: logits -> sigmoid -> top-4 (biased) -> normalized weights
// ---------------------------------------------------------------------------
__global__ __launch_bounds__(128) void k_gate(const float* __restrict__ h2,
                                              const float* __restrict__ gw,
                                              const float* __restrict__ gbias,
                                              int* __restrict__ tki,
                                              float* __restrict__ twn) {
  const int t = blockIdx.x, tid = threadIdx.x;
  const int e = tid >> 2, j = tid & 3;
  const float* x = h2 + (size_t)t * HID;
  float s = 0.f;
  for (int k = j; k < HID; k += 4) s += x[k] * gw[(size_t)k * N_EXP + e];
  s += __shfl_xor(s, 1, 32);
  s += __shfl_xor(s, 2, 32);
  __shared__ float sc[N_EXP], sb[N_EXP];
  if (j == 0) {
    float sig = 1.f / (1.f + __expf(-s));
    sc[e] = sig;
    sb[e] = sig + gbias[e];
  }
  __syncthreads();
  if (tid == 0) {
    int idx[TOP_K]; float w[TOP_K]; float wsum = 0.f;
    for (int s4 = 0; s4 < TOP_K; ++s4) {
      float best = -1e30f; int bi = 0;
      for (int e2 = 0; e2 < N_EXP; ++e2)
        if (sb[e2] > best) { best = sb[e2]; bi = e2; }
      sb[bi] = -1e30f;
      idx[s4] = bi; w[s4] = sc[bi]; wsum += sc[bi];
    }
    for (int s4 = 0; s4 < TOP_K; ++s4) {
      tki[t * TOP_K + s4] = idx[s4];
      twn[t * TOP_K + s4] = w[s4] / wsum;
    }
  }
}

// ---------------------------------------------------------------------------
// Deterministic token->expert compaction (stable token-ascending order,
// matching the reference's stable argsort + capacity drop). 1 wave / expert.
// ---------------------------------------------------------------------------
__global__ __launch_bounds__(32) void k_route(const int* __restrict__ tki,
                                              const float* __restrict__ twn,
                                              int* __restrict__ tokens,
                                              float* __restrict__ wts,
                                              int* __restrict__ cnt,
                                              int* __restrict__ tokpos) {
  const int e = blockIdx.x, lane = threadIdx.x;
  int base = 0;
  for (int s0 = 0; s0 < T_TOK * TOP_K; s0 += 32) {
    int s = s0 + lane;
    bool match = (tki[s] == e);
    unsigned m32 = (unsigned)__ballot(match);
    int prefix = __popc(m32 & ((1u << lane) - 1));
    if (match) {
      int pos = base + prefix;
      tokpos[s] = pos;
      if (pos < CAP) {
        tokens[e * CAP + pos] = s >> 2;   // token id
        wts[e * CAP + pos] = twn[s];
      }
    }
    base += __popc(m32);
  }
  if (lane == 0) cnt[e] = (base < CAP) ? base : CAP;
}

// ---------------------------------------------------------------------------
// Expert FFN1: gathered A rows; fused gate/up GEMM + SiLU; hmid bf16 out.
// Register-pipelined staging of both the gate and up weight strips.
// ---------------------------------------------------------------------------
__global__ __launch_bounds__(256) void k_moe_ffn1(const __bf16* __restrict__ h2,
                                                  const float* __restrict__ w1,
                                                  __bf16* __restrict__ hmid,
                                                  const int* __restrict__ tokens,
                                                  const int* __restrict__ cnt) {
  const int e = blockIdx.z;
  const int m0 = blockIdx.y * 64;
  const int n0 = blockIdx.x * 128;
  const int ce = cnt[e];
  if (m0 >= ce) return;
  const float* Be = w1 + (size_t)e * HID * (2 * FF);
  __shared__ alignas(16) __bf16 sG[128 * 48];
  __shared__ alignas(16) __bf16 sU[128 * 48];
  const int tid = threadIdx.x;
  const int wave = tid >> 5, lane = tid & 31;
  const int h = lane >> 4, ln = lane & 15;
  const int wm = (wave & 3) * 16;
  const int wn = (wave >> 2) * 64;
  int arow = m0 + wm + ln;
  int asrc = (arow < ce) ? tokens[e * CAP + arow] : 0;
  const __bf16* Ap = h2 + (size_t)asrc * HID;
  const float* BkG = Be + n0;
  const float* BkU = Be + n0 + FF;
  const int off0 = (tid >> 7) * 2 * (2 * FF) + (tid & 127);
  StageRegs rg, ru;
  stage_load<2 * FF>(rg, BkG, off0);
  stage_load<2 * FF>(ru, BkU, off0);
  f32x8 accG[4] = {}, accU[4] = {};
  for (int k0 = 0; k0 < HID; k0 += 32) {
    __syncthreads();
    stage_store(rg, sG, tid);
    stage_store(ru, sU, tid);
    __syncthreads();
    BkG += 32 * (2 * FF);
    BkU += 32 * (2 * FF);
    if (k0 + 32 < HID) {
      stage_load<2 * FF>(rg, BkG, off0);
      stage_load<2 * FF>(ru, BkU, off0);
    }
    bf16x16 af = load_frag(Ap, 0, h);
    Ap += 32;
#pragma unroll
    for (int ns = 0; ns < 4; ++ns) {
      bf16x16 bg = read_b_lds(sG, wn + ns * 16 + ln, h);
      accG[ns] = WMMA_BF16(af, bg, accG[ns]);
      bf16x16 bu = read_b_lds(sU, wn + ns * 16 + ln, h);
      accU[ns] = WMMA_BF16(af, bu, accU[ns]);
    }
  }
#pragma unroll
  for (int ns = 0; ns < 4; ++ns) {
    int col = n0 + wn + ns * 16 + ln;
#pragma unroll
    for (int r = 0; r < 8; ++r) {
      int row = m0 + wm + r + 8 * h;
      float g = accG[ns][r], u = accU[ns][r];
      float silu = g / (1.f + __expf(-g));
      hmid[((size_t)e * CAP + row) * FF + col] = (__bf16)(silu * u);
    }
  }
}

// ---------------------------------------------------------------------------
// Expert FFN2: eo[e] = hmid[e] (bf16) x w2[e] (f32), plain f32 out.
// ---------------------------------------------------------------------------
__global__ __launch_bounds__(256) void k_moe_ffn2(const __bf16* __restrict__ hmid,
                                                  const float* __restrict__ w2,
                                                  float* __restrict__ eo,
                                                  const int* __restrict__ cnt) {
  const int e = blockIdx.z;
  const int m0 = blockIdx.y * 64;
  const int n0 = blockIdx.x * 128;
  if (m0 >= cnt[e]) return;
  const float* Be = w2 + (size_t)e * FF * HID;
  const __bf16* Ae = hmid + (size_t)e * CAP * FF;
  float* Ce = eo + (size_t)e * CAP * HID;
  __shared__ alignas(16) __bf16 sB[128 * 48];
  const int tid = threadIdx.x;
  const int wave = tid >> 5, lane = tid & 31;
  const int h = lane >> 4, ln = lane & 15;
  const int wm = (wave & 3) * 16;
  const int wn = (wave >> 2) * 64;
  const __bf16* Ap = Ae + (size_t)(m0 + wm + ln) * FF;
  const float* Bk = Be + n0;
  const int off0 = (tid >> 7) * 2 * HID + (tid & 127);
  StageRegs sr;
  stage_load<HID>(sr, Bk, off0);
  f32x8 acc[4] = {};
  for (int k0 = 0; k0 < FF; k0 += 32) {
    __syncthreads();
    stage_store(sr, sB, tid);
    __syncthreads();
    Bk += 32 * HID;
    if (k0 + 32 < FF) stage_load<HID>(sr, Bk, off0);
    bf16x16 af = load_frag(Ap, 0, h);
    Ap += 32;
#pragma unroll
    for (int ns = 0; ns < 4; ++ns) {
      bf16x16 bf = read_b_lds(sB, wn + ns * 16 + ln, h);
      acc[ns] = WMMA_BF16(af, bf, acc[ns]);
    }
  }
#pragma unroll
  for (int ns = 0; ns < 4; ++ns) {
    int col = n0 + wn + ns * 16 + ln;
#pragma unroll
    for (int r = 0; r < 8; ++r) {
      int row = m0 + wm + r + 8 * h;
      Ce[(size_t)row * HID + col] = acc[ns][r];
    }
  }
}

// ---------------------------------------------------------------------------
// Final gather: out[t] = sum_s tw[t][s] * eo[expert][pos] — deterministic.
// ---------------------------------------------------------------------------
__global__ __launch_bounds__(256) void k_gather(const float* __restrict__ eo,
                                                const int* __restrict__ tki,
                                                const int* __restrict__ tokpos,
                                                const float* __restrict__ twn,
                                                float* __restrict__ out) {
  const int t = blockIdx.x;
  for (int c = threadIdx.x; c < HID; c += 256) {
    float acc = 0.f;
#pragma unroll
    for (int s = 0; s < TOP_K; ++s) {
      int e = tki[t * TOP_K + s];
      int p = tokpos[t * TOP_K + s];
      if (p < CAP)
        acc += twn[t * TOP_K + s] * eo[((size_t)e * CAP + p) * HID + c];
    }
    out[(size_t)t * HID + c] = acc;
  }
}

// ---------------------------------------------------------------------------
extern "C" void kernel_launch(void* const* d_in, const int* in_sizes, int n_in,
                              void* d_out, int out_size, void* d_ws, size_t ws_size,
                              hipStream_t stream) {
  (void)in_sizes; (void)n_in; (void)out_size; (void)ws_size;
  const int*   positions = (const int*)  d_in[0];
  const float* hidden    = (const float*)d_in[1];
  const float* residual  = (const float*)d_in[2];
  const float* w_qkv     = (const float*)d_in[3];
  const float* w_out     = (const float*)d_in[4];
  const float* q_ln_w    = (const float*)d_in[5];
  const float* k_ln_w    = (const float*)d_in[6];
  const float* op_norm_w = (const float*)d_in[7];
  const float* ffn_norm_w= (const float*)d_in[8];
  const float* gate_w    = (const float*)d_in[9];
  const float* gate_bias = (const float*)d_in[10];
  const float* w1        = (const float*)d_in[11];
  const float* w2        = (const float*)d_in[12];

  float* out  = (float*)d_out;                       // output 0: MoE out
  float* res2 = (float*)d_out + (size_t)T_TOK * HID; // output 1: residual2

  char* wp = (char*)d_ws;
  auto alloc = [&](size_t bytes) -> void* {
    void* p = (void*)wp;
    wp += (bytes + 255) & ~(size_t)255;
    return p;
  };
  float*  res1   = (float*) alloc(sizeof(float) * T_TOK * HID);
  __bf16* h1bf   = (__bf16*)alloc(2ull * T_TOK * HID);
  float*  qkv    = (float*) alloc(sizeof(float) * T_TOK * QKV_N);
  __bf16* Qb     = (__bf16*)alloc(2ull * N_HEADS * T_TOK * HEAD_DIM);
  __bf16* Kb     = (__bf16*)alloc(2ull * N_KV * T_TOK * HEAD_DIM);
  __bf16* Vtb    = (__bf16*)alloc(2ull * N_KV * HEAD_DIM * T_TOK);
  __bf16* Abf    = (__bf16*)alloc(2ull * T_TOK * HID);
  float*  attn   = (float*) alloc(sizeof(float) * T_TOK * HID);
  __bf16* h2bf   = (__bf16*)alloc(2ull * T_TOK * HID);
  float*  h2f    = (float*) alloc(sizeof(float) * T_TOK * HID);
  int*    tki    = (int*)   alloc(sizeof(int) * T_TOK * TOP_K);
  float*  twn    = (float*) alloc(sizeof(float) * T_TOK * TOP_K);
  int*    tokens = (int*)   alloc(sizeof(int) * N_EXP * CAP);
  float*  wts    = (float*) alloc(sizeof(float) * N_EXP * CAP);
  int*    cnt    = (int*)   alloc(sizeof(int) * N_EXP);
  int*    tokpos = (int*)   alloc(sizeof(int) * T_TOK * TOP_K);
  __bf16* hmid   = (__bf16*)alloc(2ull * N_EXP * CAP * FF);
  float*  eo     = (float*) alloc(sizeof(float) * N_EXP * CAP * HID);
  (void)wts;

  k_addnorm<<<T_TOK, 256, 0, stream>>>(hidden, residual, op_norm_w, res1, h1bf, nullptr);
  k_gemm<QKV_N, HID><<<dim3(QKV_N / 128, T_TOK / 64), 256, 0, stream>>>(h1bf, w_qkv, qkv);
  k_rope<<<T_TOK, 256, 0, stream>>>(qkv, q_ln_w, k_ln_w, positions, Qb, Kb, Vtb);
  k_attn<<<dim3(T_TOK / 16 / 8, N_HEADS), 256, 0, stream>>>(Qb, Kb, Vtb, Abf);
  k_gemm<HID, HID><<<dim3(HID / 128, T_TOK / 64), 256, 0, stream>>>(Abf, w_out, attn);
  k_addnorm<<<T_TOK, 256, 0, stream>>>(attn, res1, ffn_norm_w, res2, h2bf, h2f);
  k_gate<<<T_TOK, 128, 0, stream>>>(h2f, gate_w, gate_bias, tki, twn);
  k_route<<<N_EXP, 32, 0, stream>>>(tki, twn, tokens, wts, cnt, tokpos);
  k_moe_ffn1<<<dim3(FF / 128, CAP / 64, N_EXP), 256, 0, stream>>>(h2bf, w1, hmid, tokens, cnt);
  k_moe_ffn2<<<dim3(HID / 128, CAP / 64, N_EXP), 256, 0, stream>>>(hmid, w2, eo, cnt);
  k_gather<<<T_TOK, 256, 0, stream>>>(eo, tki, tokpos, twn, out);
}
